// SALayer_16578573763168
// MI455X (gfx1250) — matrix operand order repaired and checked by reading, not compiled, not run on hardware
//
#include <hip/hip_runtime.h>
#include <hip/hip_bf16.h>

typedef __attribute__((ext_vector_type(16))) _Float16 v16h;
typedef __attribute__((ext_vector_type(8)))  _Float16 h8;
typedef __attribute__((ext_vector_type(8)))  float    v8f;
typedef __attribute__((ext_vector_type(4)))  float    f4;
typedef __attribute__((ext_vector_type(4)))  int      i4;

#define BS    2
#define CDIM  256
#define NSEQ  4096
#define NHEAD 8
#define DHEAD 32
#define NEGM  (-1.0e30f)

#define AS1 __attribute__((address_space(1)))
#define AS3 __attribute__((address_space(3)))

#if defined(__has_builtin)
#if __has_builtin(__builtin_amdgcn_global_load_async_to_lds_b128)
#define USE_ASYNC_LDS 1
#endif
#endif
#ifndef USE_ASYNC_LDS
#define USE_ASYNC_LDS 0
#endif

static __device__ __forceinline__ void wait_asynccnt0() {
#if defined(__has_builtin) && __has_builtin(__builtin_amdgcn_s_wait_asynccnt)
  __builtin_amdgcn_s_wait_asynccnt(0);
#else
  asm volatile("s_wait_asynccnt 0" ::: "memory");
#endif
}

static __device__ __forceinline__ v8f wmma16(v16h a, v16h b, v8f c) {
  // v_wmma_f32_16x16x32_f16: D = A(16x32 f16) * B(32x16 f16) + C(16x16 f32)
  return __builtin_amdgcn_wmma_f32_16x16x32_f16(false, a, false, b, (short)0, c,
                                                false, false);
}

// 16-byte per-lane copy global -> LDS (async on gfx1250 toolchains that expose it)
static __device__ __forceinline__ void copy16_to_lds(const _Float16* g,
                                                     _Float16* l) {
#if USE_ASYNC_LDS
  __builtin_amdgcn_global_load_async_to_lds_b128((AS1 i4*)g, (AS3 i4*)l, 0, 0);
#else
  *(h8*)l = *(const h8*)g;
#endif
}

// ---------------------------------------------------------------------------
// Kernel 1: QKV projection.  y = W @ x + b  (per batch, per matrix)
//   x: (bs, C, N) f32.   W: (C, C) f32.   b: (C,) f32.
// Outputs (f16):
//   Kt, Qt : (bs, H, N, 32)  -- transposed, so attention A/B frags are contiguous
//   Vh     : (bs, C, N)
// grid = (N/16, bs*3), block = 256 (8 waves; each wave 2 channel-tiles of 16)
// ---------------------------------------------------------------------------
__global__ __launch_bounds__(256) void qkv_proj_kernel(
    const float* __restrict__ feats,
    const float* __restrict__ Wk, const float* __restrict__ bk,
    const float* __restrict__ Wq, const float* __restrict__ bq,
    const float* __restrict__ Wv, const float* __restrict__ bv,
    _Float16* __restrict__ Kt, _Float16* __restrict__ Qt,
    _Float16* __restrict__ Vh)
{
  // x tile (256 channels x 16 positions) transposed to f16, padded stride 264
  __shared__ _Float16 xT[16 * 264];

  const int t    = threadIdx.x;
  const int nb   = blockIdx.x;              // n-tile index
  const int bm   = blockIdx.y;              // b*3 + mat
  const int b    = bm / 3;
  const int mat  = bm % 3;                  // 0=K, 1=Q, 2=V
  const int n0   = nb * 16;

  const float* Wm   = (mat == 0) ? Wk : ((mat == 1) ? Wq : Wv);
  const float* bias = (mat == 0) ? bk : ((mat == 1) ? bq : bv);
  const float* xb   = feats + (size_t)b * CDIM * NSEQ;

  // Cooperative load: 256x16 f32 tile -> LDS transposed f16 (16 x 264)
#pragma unroll
  for (int it = 0; it < 4; ++it) {
    int p  = it * 256 + t;                  // float4 id, 1024 total
    int r  = p >> 2;                        // input channel 0..255
    int c4 = (p & 3) * 4;                   // tile col base
    f4 v = *(const f4*)(xb + (size_t)r * NSEQ + n0 + c4);
    xT[(c4 + 0) * 264 + r] = (_Float16)v.x;
    xT[(c4 + 1) * 264 + r] = (_Float16)v.y;
    xT[(c4 + 2) * 264 + r] = (_Float16)v.z;
    xT[(c4 + 3) * 264 + r] = (_Float16)v.w;
  }
  __syncthreads();

  const int lane = t & 31;
  const int wave = t >> 5;
  const int col  = lane & 15;               // N-lane of B/C/D frags
  const int grp  = lane >> 4;               // lane half-group

  v8f acc0 = {};
  v8f acc1 = {};

  for (int kb = 0; kb < CDIM; kb += 32) {
    // B fragment from LDS: b[j<8] = x[kb + 8*grp + j][n0 + col], b[j>=8] = x[kb + 16 + 8*grp + (j-8)][n0 + col]
    v16h bt;
    {
      const _Float16* src = &xT[col * 264 + kb + 8 * grp];
      h8 lo = *(const h8*)(src);
      h8 hi = *(const h8*)(src + 16);
#pragma unroll
      for (int j = 0; j < 8; ++j) { bt[j] = lo[j]; bt[8 + j] = hi[j]; }
    }
#pragma unroll
    for (int tt = 0; tt < 2; ++tt) {
      // A fragment: a[j<8] = W[o][kb+8*grp+j], a[j>=8] = W[o][kb+16+8*grp+(j-8)]
      int o = wave * 32 + tt * 16 + col;
      const float* wrow = Wm + (size_t)o * CDIM + kb;
      f4 w0 = *(const f4*)(wrow + 8 * grp);
      f4 w1 = *(const f4*)(wrow + 8 * grp + 4);
      f4 w2 = *(const f4*)(wrow + 16 + 8 * grp);
      f4 w3 = *(const f4*)(wrow + 16 + 8 * grp + 4);
      v16h at;
      at[0]  = (_Float16)w0.x; at[1]  = (_Float16)w0.y;
      at[2]  = (_Float16)w0.z; at[3]  = (_Float16)w0.w;
      at[4]  = (_Float16)w1.x; at[5]  = (_Float16)w1.y;
      at[6]  = (_Float16)w1.z; at[7]  = (_Float16)w1.w;
      at[8]  = (_Float16)w2.x; at[9]  = (_Float16)w2.y;
      at[10] = (_Float16)w2.z; at[11] = (_Float16)w2.w;
      at[12] = (_Float16)w3.x; at[13] = (_Float16)w3.y;
      at[14] = (_Float16)w3.z; at[15] = (_Float16)w3.w;
      if (tt == 0) acc0 = wmma16(at, bt, acc0);
      else         acc1 = wmma16(at, bt, acc1);
    }
  }

  // Store with bias. C/D layout: lane holds M = v + 8*grp, N = col.
  const int nidx = n0 + col;
#pragma unroll
  for (int tt = 0; tt < 2; ++tt) {
#pragma unroll
    for (int v = 0; v < 8; ++v) {
      int o = wave * 32 + tt * 16 + v + 8 * grp;
      float val = (tt == 0 ? acc0[v] : acc1[v]) + bias[o];
      _Float16 hv = (_Float16)val;
      if (mat == 2) {
        Vh[((size_t)b * CDIM + o) * NSEQ + nidx] = hv;
      } else {
        _Float16* dst = (mat == 0) ? Kt : Qt;
        int hh = o >> 5, cl = o & 31;
        dst[(((size_t)b * NHEAD + hh) * NSEQ + nidx) * DHEAD + cl] = hv;
      }
    }
  }
}

// ---------------------------------------------------------------------------
// Kernel 2: flash attention with softmax over the *query* axis m.
// Per output row n: weights over m from K[:,n].Q[:,m]; out = sum_m w_m V[:,m].
// A block = 8 waves sharing one (b,h); each wave owns a 16-row n-tile.
// Q/V m-chunks of 64 are staged into double-buffered LDS with async copies,
// overlapping the next chunk's global->LDS DMA with this chunk's WMMAs.
// grid = 512 (16 bh * 32 tile-groups), block = 256.
// ---------------------------------------------------------------------------
__global__ __launch_bounds__(256) void flash_attn_kernel(
    const _Float16* __restrict__ Kt, const _Float16* __restrict__ Qt,
    const _Float16* __restrict__ Vh, const float* __restrict__ feats,
    const int* __restrict__ masks, float* __restrict__ out)
{
  __shared__ _Float16 qs[2][64 * 40];       // Q chunk: 64 m-rows x 32 ch (pad 40)
  __shared__ _Float16 vsh[2][32 * 72];      // V chunk: 32 c-rows x 64 m (pad 72)
  __shared__ _Float16 pbuf[8][16 * 72];     // per-wave P staging (pad 72)

  const int t    = threadIdx.x;
  const int lane = t & 31;
  const int wave = t >> 5;
  const int bh   = blockIdx.x >> 5;         // (b*H + h)
  const int tg   = blockIdx.x & 31;         // tile-group
  const int b    = bh >> 3;
  const int h    = bh & 7;
  const int col  = lane & 15;
  const int grp  = lane >> 4;
  const int n0   = (tg * 8 + wave) * 16;

  const float scale = 0.17677669529663687f; // 1/sqrt(32)

  const _Float16* qbase = Qt + ((size_t)bh) * NSEQ * DHEAD;
  const _Float16* vbase = Vh + ((size_t)b * CDIM + h * DHEAD) * NSEQ;
  const int* mrow = masks + (size_t)b * NSEQ;

  // cooperative-load index mapping (256 threads, 16B each)
  const int qrow = t >> 2, qch = t & 3;     // Q: 64 rows x 4 chunks of 8 halves
  const int vrow = t >> 3, vch = t & 7;     // V: 32 rows x 8 chunks of 8 halves

  // stage chunk m0 into buffer bi (one b128 per thread per matrix)
  auto stage = [&](int m0, int bi) {
    copy16_to_lds(qbase + (size_t)(m0 + qrow) * DHEAD + qch * 8,
                  &qs[bi][qrow * 40 + qch * 8]);
    copy16_to_lds(vbase + (size_t)vrow * NSEQ + m0 + vch * 8,
                  &vsh[bi][vrow * 72 + vch * 8]);
  };

  // A fragment (constant over m loop), scale folded into K:
  // a[j] = scale * K^T[n0+col][chan(j)]   (16x32 f16)
  v16h a;
  {
    const _Float16* kp = Kt + (((size_t)bh) * NSEQ + n0 + col) * DHEAD;
    h8 lo = *(const h8*)(kp + 16 * grp);
    h8 hi = *(const h8*)(kp + 16 * grp + 8);
#pragma unroll
    for (int j = 0; j < 8; ++j) {
      a[j]     = (_Float16)((float)lo[j] * scale);
      a[8 + j] = (_Float16)((float)hi[j] * scale);
    }
  }

  v8f o0 = {};
  v8f o1 = {};
  float mi[8], li[8];
#pragma unroll
  for (int v = 0; v < 8; ++v) { mi[v] = -3.0e38f; li[v] = 0.0f; }

  _Float16* pw = &pbuf[wave][0];

  // prologue: stage chunk 0
  stage(0, 0);
#if USE_ASYNC_LDS
  wait_asynccnt0();
#endif
  __syncthreads();

  for (int m0 = 0; m0 < NSEQ; m0 += 64) {
    const int cur = (m0 >> 6) & 1;

    // kick off the next chunk's global->LDS copy (overlaps with compute below)
    if (m0 + 64 < NSEQ) stage(m0 + 64, cur ^ 1);

    const _Float16* qsc = &qs[cur][0];
    const _Float16* vsc = &vsh[cur][0];

    // ---- S = (scale*K)^T Q for 16(n) x 64(m): four WMMAs, K=32=dh
    v8f s0 = {}, s1 = {}, s2 = {}, s3 = {};
#pragma unroll
    for (int si = 0; si < 4; ++si) {
      v16h bq;
      const _Float16* qp = qsc + (16 * si + col) * 40 + 16 * grp;
      h8 x0 = *(const h8*)qp;
      h8 x1 = *(const h8*)(qp + 8);
#pragma unroll
      for (int j = 0; j < 8; ++j) { bq[j] = x0[j]; bq[8 + j] = x1[j]; }
      if      (si == 0) s0 = wmma16(a, bq, s0);
      else if (si == 1) s1 = wmma16(a, bq, s1);
      else if (si == 2) s2 = wmma16(a, bq, s2);
      else              s3 = wmma16(a, bq, s3);
    }

    // ---- mask (along m == lane column)
    const int mk0 = mrow[m0 + col];
    const int mk1 = mrow[m0 + 16 + col];
    const int mk2 = mrow[m0 + 32 + col];
    const int mk3 = mrow[m0 + 48 + col];

    // ---- online softmax; row stats mapping matches C/D accumulator layout
#pragma unroll
    for (int v = 0; v < 8; ++v) {
      float x0 = mk0 ? s0[v] : NEGM;
      float x1 = mk1 ? s1[v] : NEGM;
      float x2 = mk2 ? s2[v] : NEGM;
      float x3 = mk3 ? s3[v] : NEGM;
      float xm = fmaxf(fmaxf(x0, x1), fmaxf(x2, x3));
      xm = fmaxf(xm, __shfl_xor(xm, 1));
      xm = fmaxf(xm, __shfl_xor(xm, 2));
      xm = fmaxf(xm, __shfl_xor(xm, 4));
      xm = fmaxf(xm, __shfl_xor(xm, 8));
      float mnew = fmaxf(mi[v], xm);
      float al = __expf(mi[v] - mnew);
      float p0 = __expf(x0 - mnew);
      float p1 = __expf(x1 - mnew);
      float p2 = __expf(x2 - mnew);
      float p3 = __expf(x3 - mnew);
      float rs = (p0 + p1) + (p2 + p3);
      rs += __shfl_xor(rs, 1);
      rs += __shfl_xor(rs, 2);
      rs += __shfl_xor(rs, 4);
      rs += __shfl_xor(rs, 8);
      li[v] = li[v] * al + rs;
      mi[v] = mnew;
      o0[v] *= al;
      o1[v] *= al;
      int r = v + 8 * grp;                  // row in n-tile
      pw[r * 72 + col]      = (_Float16)p0;
      pw[r * 72 + 16 + col] = (_Float16)p1;
      pw[r * 72 + 32 + col] = (_Float16)p2;
      pw[r * 72 + 48 + col] = (_Float16)p3;
    }

    // ---- O^T += P * V^T : two K=32 sub-chunks x two 16-col c tiles
#pragma unroll
    for (int kk = 0; kk < 64; kk += 32) {
      v16h pa;
      {
        const _Float16* pr = pw + col * 72 + kk;
        h8 l0 = *(const h8*)(pr + 16 * grp);
        h8 l1 = *(const h8*)(pr + 16 * grp + 8);
#pragma unroll
        for (int j = 0; j < 8; ++j) { pa[j] = l0[j]; pa[8 + j] = l1[j]; }
      }
      v16h bv0, bv1;
      {
        const _Float16* vp0 = vsc + col * 72 + kk + 16 * grp;
        h8 x0 = *(const h8*)vp0;
        h8 x1 = *(const h8*)(vp0 + 8);
        const _Float16* vp1 = vsc + (16 + col) * 72 + kk + 16 * grp;
        h8 y0 = *(const h8*)vp1;
        h8 y1 = *(const h8*)(vp1 + 8);
#pragma unroll
        for (int j = 0; j < 8; ++j) {
          bv0[j] = x0[j]; bv0[8 + j] = x1[j];
          bv1[j] = y0[j]; bv1[8 + j] = y1[j];
        }
      }
      o0 = wmma16(pa, bv0, o0);
      o1 = wmma16(pa, bv1, o1);
    }

    // next buffer must be fully written and everyone done reading `cur`
#if USE_ASYNC_LDS
    wait_asynccnt0();
#endif
    __syncthreads();
  }

  // ---- epilogue: divide by row sum, add residual, store fp32
  const float* fb = feats + ((size_t)b * CDIM + h * DHEAD) * NSEQ;
  float*       ob = out   + ((size_t)b * CDIM + h * DHEAD) * NSEQ;
#pragma unroll
  for (int v = 0; v < 8; ++v) {
    float inv = 1.0f / fmaxf(li[v], 1e-30f);
    int nidx = n0 + v + 8 * grp;
    size_t i0 = (size_t)col * NSEQ + nidx;
    size_t i1 = (size_t)(16 + col) * NSEQ + nidx;
    ob[i0] = o0[v] * inv + fb[i0];
    ob[i1] = o1[v] * inv + fb[i1];
  }
}

// ---------------------------------------------------------------------------
extern "C" void kernel_launch(void* const* d_in, const int* in_sizes, int n_in,
                              void* d_out, int out_size, void* d_ws, size_t ws_size,
                              hipStream_t stream) {
  const float* feats = (const float*)d_in[0];
  const int*   masks = (const int*)d_in[1];
  const float* Wk    = (const float*)d_in[2];
  const float* bk    = (const float*)d_in[3];
  const float* Wq    = (const float*)d_in[4];
  const float* bq    = (const float*)d_in[5];
  const float* Wv    = (const float*)d_in[6];
  const float* bv    = (const float*)d_in[7];
  float* out = (float*)d_out;

  // workspace: Kt(4MB) | Qt(4MB) | Vh(4MB), all f16
  const size_t kt_elems = (size_t)BS * NHEAD * NSEQ * DHEAD;
  _Float16* Kt = (_Float16*)d_ws;
  _Float16* Qt = Kt + kt_elems;
  _Float16* Vh = Qt + kt_elems;

  dim3 g1(NSEQ / 16, BS * 3);
  qkv_proj_kernel<<<g1, 256, 0, stream>>>(feats, Wk, bk, Wq, bq, Wv, bv,
                                          Kt, Qt, Vh);

  dim3 g2(BS * NHEAD * 32);                 // 512 blocks, 8 waves each
  flash_attn_kernel<<<g2, 256, 0, stream>>>(Kt, Qt, Vh, feats, masks, out);
}
